// MoEResKANet_687194768034
// MI455X (gfx1250) — compile-verified, matrix operand order checked by reading, not run on hardware
//
#include <hip/hip_runtime.h>

// ---------------------------------------------------------------------------
// MoE-KAN ResNet forward for gfx1250 (MI455X).
// Convs: implicit-GEMM, v_wmma_f32_16x16x32_bf16, 64x16 tile per wave
// (4 accumulators / 4 WMMAs per K-step, shared B fragment).
// Activations pre-transformed (SiLU / Legendre basis) to bf16 once per use;
// weights pre-packed to bf16 with K padded to a multiple of 32.
// Weight tiles stream into LDS via GLOBAL_LOAD_ASYNC_TO_LDS_B128 (ASYNCcnt
// path) — builtin signature: (global v4i*, lds v4i*, imm offset, imm cpol).
// ---------------------------------------------------------------------------

typedef __attribute__((ext_vector_type(16))) __bf16 v16bf;
typedef __attribute__((ext_vector_type(8)))  float  v8f;
typedef int v4i_async __attribute__((vector_size(16)));

#define TK 32

#if __has_builtin(__builtin_amdgcn_global_load_async_to_lds_b128) && \
    __has_builtin(__builtin_amdgcn_s_wait_asynccnt)
#define HAVE_ASYNC_LDS 1
#endif

__device__ __forceinline__ float siluf(float v) {
    return v / (1.0f + __expf(-v));
}

// ---------------------------------------------------------------------------
// Implicit-GEMM conv: A = im2col(bf16 activation tensor, CC channels),
// B = packed bf16 weights [Cout][Kp].  64 rows x 16 cols per wave.
// ---------------------------------------------------------------------------
__global__ __launch_bounds__(32)
void k_conv_wmma(const __bf16* __restrict__ xb, const __bf16* __restrict__ wb,
                 float* __restrict__ y,
                 int Bn, int CC, int H, int W, int Cout, int OH, int OW,
                 int KS, int S, int P, int Kdim, int Kp, int beta)
{
    __shared__ __align__(16) __bf16 As[4][16][TK];
    __shared__ __align__(16) __bf16 Bs[16][TK];       // [co][k]

    const int lane = threadIdx.x;                      // wave32
    const int m0 = blockIdx.x * 64;
    const int n0 = blockIdx.y * 16;
    const int M  = Bn * OH * OW;
    const int KK = KS * KS;
    const int OHW = OH * OW;

    // Each lane owns two A rows (lane, lane+32): decode output coords once.
    int  rb[2], rih0[2], riw0[2];
    bool rvalid[2];
    #pragma unroll
    for (int t = 0; t < 2; ++t) {
        int row = m0 + lane + t * 32;
        rvalid[t] = row < M;
        int r  = rvalid[t] ? row : 0;
        int b  = r / OHW;
        int pix = r - b * OHW;
        int oh = pix / OW, ow = pix - oh * OW;
        rb[t] = b; rih0[t] = oh * S - P; riw0[t] = ow * S - P;
    }

    v8f acc[4] = {};

    for (int k0 = 0; k0 < Kp; k0 += TK) {
        // ---- B tile: 16 co x 32 k (weights, zero-padded, 64B rows) ----
#ifdef HAVE_ASYNC_LDS
        #pragma unroll
        for (int it = 0; it < 2; ++it) {
            int chunk = it * 32 + lane;                // 0..63: 16B chunks
            int co = chunk >> 2, seg = chunk & 3;
            const __bf16* gp = wb + (size_t)(n0 + co) * Kp + k0 + seg * 8;
            __builtin_amdgcn_global_load_async_to_lds_b128(
                (__attribute__((address_space(1))) v4i_async*)gp,
                (__attribute__((address_space(3))) v4i_async*)&Bs[co][seg * 8],
                0, 0);
        }
#else
        for (int i = lane; i < 16 * TK; i += 32) {
            int co = i >> 5, kk = i & 31;
            Bs[co][kk] = wb[(size_t)(n0 + co) * Kp + k0 + kk];
        }
#endif
        // ---- A tiles: 64 rows x 32 k, incremental (cc,kh,kw) decode ----
        #pragma unroll
        for (int t = 0; t < 2; ++t) {
            int r = lane + t * 32;
            __bf16* dst = &As[r >> 4][r & 15][0];
            if (!rvalid[t]) {
                for (int kk = 0; kk < TK; ++kk) dst[kk] = (__bf16)0.f;
            } else {
                int k  = k0;
                int cc = k / KK;
                int rem = k - cc * KK;
                int kh = rem / KS, kw = rem - kh * KS;
                const __bf16* xbase = xb + (size_t)rb[t] * CC * H * W;
                const int ih0 = rih0[t], iw0 = riw0[t];
                for (int kk = 0; kk < TK; ++kk) {
                    float v = 0.f;
                    if (k + kk < Kdim) {
                        int ih = ih0 + kh, iw = iw0 + kw;
                        if (ih >= 0 && ih < H && iw >= 0 && iw < W)
                            v = (float)xbase[((size_t)cc * H + ih) * W + iw];
                    }
                    dst[kk] = (__bf16)v;
                    if (++kw == KS) { kw = 0; if (++kh == KS) { kh = 0; ++cc; } }
                }
            }
        }
        if (k0 + TK < Kp && lane == 0)
            __builtin_prefetch(wb + (size_t)n0 * Kp + k0 + TK, 0, 1);
#ifdef HAVE_ASYNC_LDS
        __builtin_amdgcn_s_wait_asynccnt(0);
#endif
        __syncthreads();

        // ---- fragments (16-bit A 16x32 layout; B mirrored; N = lane%16) ----
        const int hf = lane >> 4, ml = lane & 15;
        v16bf bfrag;
        #pragma unroll
        for (int e = 0; e < 16; ++e) {
            int kk = (e >> 3) * 16 + hf * 8 + (e & 7);
            bfrag[e] = Bs[ml][kk];
        }
        #pragma unroll
        for (int t = 0; t < 4; ++t) {
            v16bf afrag;
            #pragma unroll
            for (int e = 0; e < 16; ++e) {
                int kk = (e >> 3) * 16 + hf * 8 + (e & 7);
                afrag[e] = As[t][ml][kk];
            }
            acc[t] = __builtin_amdgcn_wmma_f32_16x16x32_bf16(
                         false, afrag, false, bfrag, (short)0, acc[t], false, false);
        }
        __syncthreads();
    }

    // ---- C store: VGPR r -> M = t*16 + (lane/16)*8 + r, N = lane%16 ----
    const int hf = lane >> 4, nl = lane & 15;
    const int co = n0 + nl;                    // Cout is a multiple of 16
    #pragma unroll
    for (int t = 0; t < 4; ++t) {
        #pragma unroll
        for (int r2 = 0; r2 < 8; ++r2) {
            int row = m0 + t * 16 + hf * 8 + r2;
            if (row < M) {
                int b = row / OHW;
                int pix = row - b * OHW;
                int oh = pix / OW, ow = pix - oh * OW;
                float* dst = &y[((size_t)(b * Cout + co) * OH + oh) * OW + ow];
                float v = acc[t][r2];
                *dst = beta ? (*dst + v) : v;
            }
        }
    }
}

// ---------------------------------------------------------------------------
// Pre-transform / packing kernels
// ---------------------------------------------------------------------------

// weights fp32 [Cout][Kdim] -> bf16 [Cout][Kp], zero-padded
__global__ void k_pack_w(const float* __restrict__ w, __bf16* __restrict__ wb,
                         int Cout, int Kdim, int Kp) {
    int i = blockIdx.x * blockDim.x + threadIdx.x;
    int n = Cout * Kp;
    if (i < n) {
        int co = i / Kp, k = i - co * Kp;
        float v = (k < Kdim) ? w[(size_t)co * Kdim + k] : 0.f;
        wb[i] = (__bf16)v;
    }
}

// bf16(silu(x))
__global__ void k_prep_base(const float* __restrict__ x, __bf16* __restrict__ o, int n) {
    int i = blockIdx.x * blockDim.x + threadIdx.x;
    if (i < n) o[i] = (__bf16)siluf(x[i]);
}

// Legendre basis of minmax-normed x: [B,C,HW] -> bf16 [B,4C,HW]
__global__ void k_prep_poly(const float* __restrict__ x, __bf16* __restrict__ o,
                            const float* __restrict__ mm, int Bn, int C, int HW) {
    int i = blockIdx.x * blockDim.x + threadIdx.x;
    int n = Bn * C * HW;
    if (i >= n) return;
    int b = i / (C * HW);
    int rem = i - b * C * HW;                  // c*HW + pix
    float xmin = mm[0];
    float rng = mm[1] - xmin;
    if (!(rng > 0.f)) rng = 1.f;
    float xn = 2.f * (x[i] - xmin) / rng - 1.f;
    size_t ob = (size_t)b * 4 * C * HW + rem;
    o[ob]              = (__bf16)1.f;
    o[ob + C * HW]     = (__bf16)xn;
    o[ob + 2 * C * HW] = (__bf16)(0.5f * (3.f * xn * xn - 1.f));
    o[ob + 3 * C * HW] = (__bf16)(0.5f * (5.f * xn * xn * xn - 3.f * xn));
}

// ---------------------------------------------------------------------------
// Support kernels
// ---------------------------------------------------------------------------

__global__ void k_zero(float* p, int n) {
    int i = blockIdx.x * blockDim.x + threadIdx.x;
    if (i < n) p[i] = 0.f;
}

__global__ void k_init_loss(float* loss) {
    if (threadIdx.x == 0 && blockIdx.x == 0) *loss = 0.f;
}

// per-(b,c) mean over HW -> out[b*C+c]; one block per (b,c)
__global__ void k_chan_mean(const float* __restrict__ x, float* __restrict__ out, int HW) {
    __shared__ float s[256];
    const float* xp = x + (size_t)blockIdx.x * HW;
    float acc = 0.f;
    for (int i = threadIdx.x; i < HW; i += blockDim.x) acc += xp[i];
    s[threadIdx.x] = acc;
    __syncthreads();
    for (int o = 128; o > 0; o >>= 1) {
        if (threadIdx.x < o) s[threadIdx.x] += s[threadIdx.x + o];
        __syncthreads();
    }
    if (threadIdx.x == 0) out[blockIdx.x] = s[0] / (float)HW;
}

// whole-tensor min/max, single block of 1024 threads -> mm[0]=min, mm[1]=max
__global__ void k_global_minmax(const float* __restrict__ x, int n, float* __restrict__ mm) {
    __shared__ float smn[1024], smx[1024];
    float mn = __builtin_inff(), mx = -__builtin_inff();
    for (int i = threadIdx.x; i < n; i += blockDim.x) {
        float v = x[i];
        mn = fminf(mn, v);
        mx = fmaxf(mx, v);
    }
    smn[threadIdx.x] = mn; smx[threadIdx.x] = mx;
    __syncthreads();
    for (int o = 512; o > 0; o >>= 1) {
        if (threadIdx.x < o) {
            smn[threadIdx.x] = fminf(smn[threadIdx.x], smn[threadIdx.x + o]);
            smx[threadIdx.x] = fmaxf(smx[threadIdx.x], smx[threadIdx.x + o]);
        }
        __syncthreads();
    }
    if (threadIdx.x == 0) { mm[0] = smn[0]; mm[1] = smx[0]; }
}

// top-2 gating + cv^2 load-balancing loss (B=4, E=8)
__global__ void k_gate(const float* __restrict__ gatein, const float* __restrict__ wg,
                       int C, float* __restrict__ gates, float* __restrict__ loss) {
    if (threadIdx.x != 0 || blockIdx.x != 0) return;
    const int B = 4, E = 8;
    float logit[4][8];
    for (int b = 0; b < B; ++b)
        for (int e = 0; e < E; ++e) {
            float a = 0.f;
            for (int c = 0; c < C; ++c) a += gatein[b * C + c] * wg[c * E + e];
            logit[b][e] = a;
        }
    for (int i = 0; i < B * E; ++i) gates[i] = 0.f;
    for (int b = 0; b < B; ++b) {
        int i1 = 0;
        for (int e = 1; e < E; ++e) if (logit[b][e] > logit[b][i1]) i1 = e;
        int i2 = (i1 == 0) ? 1 : 0;
        for (int e = 0; e < E; ++e)
            if (e != i1 && logit[b][e] > logit[b][i2]) i2 = e;
        float v1 = logit[b][i1], v2 = logit[b][i2];
        float e2 = __expf(v2 - v1);
        float g1 = 1.f / (1.f + e2);
        gates[b * E + i1] = g1;
        gates[b * E + i2] = 1.f - g1;
    }
    float imp[8], load[8];
    for (int e = 0; e < E; ++e) {
        float s = 0.f, l = 0.f;
        for (int b = 0; b < B; ++b) {
            float g = gates[b * E + e];
            s += g;
            if (g > 0.f) l += 1.f;
        }
        imp[e] = s; load[e] = l;
    }
    float cv = 0.f;
    for (int t = 0; t < 2; ++t) {
        const float* v = (t == 0) ? imp : load;
        float m = 0.f;
        for (int e = 0; e < E; ++e) m += v[e];
        m /= (float)E;
        float var = 0.f;
        for (int e = 0; e < E; ++e) { float d = v[e] - m; var += d * d; }
        var /= (float)(E - 1);                       // unbiased, torch .var()
        cv += var / (m * m + 1e-10f);
    }
    *loss += 1e-2f * cv;
}

// per-expert min/max over dispatched subset; grid = E blocks
__global__ void k_expert_minmax(const float* __restrict__ x, const float* __restrict__ gates,
                                int Bn, int CHW, float* __restrict__ mmE) {
    __shared__ float smn[256], smx[256];
    const int e = blockIdx.x;
    bool sel[4];
    bool has = false;
    for (int b = 0; b < 4; ++b) { sel[b] = gates[b * 8 + e] > 0.f; has |= sel[b]; }
    float mn = __builtin_inff(), mx = -__builtin_inff();
    for (int b = 0; b < Bn; ++b) {
        if (!sel[b]) continue;
        const float* xp = x + (size_t)b * CHW;
        for (int i = threadIdx.x; i < CHW; i += blockDim.x) {
            float v = xp[i];
            mn = fminf(mn, v);
            mx = fmaxf(mx, v);
        }
    }
    smn[threadIdx.x] = mn; smx[threadIdx.x] = mx;
    __syncthreads();
    for (int o = 128; o > 0; o >>= 1) {
        if (threadIdx.x < o) {
            smn[threadIdx.x] = fminf(smn[threadIdx.x], smn[threadIdx.x + o]);
            smx[threadIdx.x] = fmaxf(smx[threadIdx.x], smx[threadIdx.x + o]);
        }
        __syncthreads();
    }
    if (threadIdx.x == 0) {
        if (has) { mmE[2 * e] = smn[0]; mmE[2 * e + 1] = smx[0]; }
        else     { mmE[2 * e] = 0.f;    mmE[2 * e + 1] = 1.f;    }
    }
}

// InstanceNorm2d(affine=False), eps=1e-5, biased variance; one block per (b,c)
__global__ void k_instnorm(float* __restrict__ t, int HW) {
    __shared__ float s1[256], s2[256];
    float* xp = t + (size_t)blockIdx.x * HW;
    float a = 0.f, b = 0.f;
    for (int i = threadIdx.x; i < HW; i += blockDim.x) {
        float v = xp[i];
        a += v; b += v * v;
    }
    s1[threadIdx.x] = a; s2[threadIdx.x] = b;
    __syncthreads();
    for (int o = 128; o > 0; o >>= 1) {
        if (threadIdx.x < o) {
            s1[threadIdx.x] += s1[threadIdx.x + o];
            s2[threadIdx.x] += s2[threadIdx.x + o];
        }
        __syncthreads();
    }
    float mean = s1[0] / (float)HW;
    float var  = s2[0] / (float)HW - mean * mean;
    float inv  = rsqrtf(var + 1e-5f);
    for (int i = threadIdx.x; i < HW; i += blockDim.x)
        xp[i] = (xp[i] - mean) * inv;
}

// y[i] += gates[b*8+e] * t[i]
__global__ void k_combine(float* __restrict__ y, const float* __restrict__ t,
                          const float* __restrict__ gates, int e, int perB, int n) {
    int i = blockIdx.x * blockDim.x + threadIdx.x;
    if (i < n) {
        int b = i / perB;
        y[i] += gates[b * 8 + e] * t[i];
    }
}

__global__ void k_add(const float* __restrict__ a, const float* __restrict__ b,
                      float* __restrict__ o, int n) {
    int i = blockIdx.x * blockDim.x + threadIdx.x;
    if (i < n) o[i] = a[i] + b[i];
}

// maxpool 3x3 stride 2 pad 1
__global__ void k_maxpool(const float* __restrict__ x, float* __restrict__ y,
                          int BC, int H, int W, int OH, int OW) {
    int i = blockIdx.x * blockDim.x + threadIdx.x;
    int n = BC * OH * OW;
    if (i >= n) return;
    int bc  = i / (OH * OW);
    int pix = i - bc * (OH * OW);
    int oh = pix / OW, ow = pix - (pix / OW) * OW;
    float m = -__builtin_inff();
    for (int kh = 0; kh < 3; ++kh)
        for (int kw = 0; kw < 3; ++kw) {
            int ih = oh * 2 - 1 + kh, iw = ow * 2 - 1 + kw;
            if (ih >= 0 && ih < H && iw >= 0 && iw < W)
                m = fmaxf(m, x[((size_t)bc * H + ih) * W + iw]);
        }
    y[i] = m;
}

// logits[b,o] = feat[b,:] . fc_w[o,:] + fc_b[o]
__global__ void k_fc(const float* __restrict__ feat, const float* __restrict__ wf,
                     const float* __restrict__ bf, float* __restrict__ out) {
    int i = blockIdx.x * blockDim.x + threadIdx.x;
    if (i >= 4 * 1000) return;
    int b = i / 1000, o = i - b * 1000;
    float a = bf[o];
    for (int k = 0; k < 128; ++k) a += feat[b * 128 + k] * wf[o * 128 + k];
    out[i] = a;
}

// ---------------------------------------------------------------------------
// Host orchestration
// ---------------------------------------------------------------------------

static inline int cdiv_(int a, int b) { return (a + b - 1) / b; }

extern "C" void kernel_launch(void* const* d_in, const int* in_sizes, int n_in,
                              void* d_out, int out_size, void* d_ws, size_t ws_size,
                              hipStream_t stream) {
    (void)in_sizes; (void)n_in; (void)out_size; (void)ws_size;

    const float* x = (const float*)d_in[0];
    const float* stem_base = (const float*)d_in[1];
    const float* stem_poly = (const float*)d_in[2];
    const float* fc_w = (const float*)d_in[31];
    const float* fc_b = (const float*)d_in[32];

    float* ws = (float*)d_ws;
    // workspace layout (float units)
    const size_t oA  = 0;               // 1,605,632 (tmp / expert conv out)
    const size_t oB  = 1605632;         // 1,605,632 (current activation)
    const size_t oY  = 3211264;         //   401,408 (MoE accumulator)
    const size_t oC2 = 3612672;         //   401,408
    const size_t oDS = 4014080;         //   401,408
    const size_t oBB = 4415488;         //   301,056 floats = 602,112 bf16 (base input)
    const size_t oPB = 4716544;         // 1,204,224 floats = 2,408,448 bf16 (poly input)
    const size_t oWB = 5920768;         //   147,456 floats = 294,912 bf16 (packed weights)
    const size_t oMM = 6068224;         //        64
    const size_t oGI = 6068288;         //       512
    const size_t oGT = 6068800;         //        32
    const size_t oFT = 6068832;         //       512

    __bf16* bb   = (__bf16*)(ws + oBB);
    __bf16* pb   = (__bf16*)(ws + oPB);
    __bf16* wbuf = (__bf16*)(ws + oWB);
    float*  loss = (float*)d_out + 4000;

    auto packw = [&](const float* w, int Cout, int Kdim) -> int {
        int Kp = (Kdim + 31) & ~31;
        int n = Cout * Kp;
        k_pack_w<<<cdiv_(n, 256), 256, 0, stream>>>(w, wbuf, Cout, Kdim, Kp);
        return Kp;
    };
    auto conv = [&](const __bf16* xin, float* yout,
                    int CC, int H, int W, int Cout, int OH, int OW,
                    int KS, int S, int P, int Kdim, int Kp, int beta) {
        dim3 g(cdiv_(4 * OH * OW, 64), Cout / 16);
        k_conv_wmma<<<g, 32, 0, stream>>>(xin, wbuf, yout, 4, CC, H, W, Cout,
                                          OH, OW, KS, S, P, Kdim, Kp, beta);
    };

    k_init_loss<<<1, 32, 0, stream>>>(loss);

    // ---- stem: KALN 7x7 s2 p3: [4,3,224,224] -> [4,32,112,112] ----
    {
        const int n = 4 * 3 * 224 * 224;
        k_global_minmax<<<1, 1024, 0, stream>>>(x, n, ws + oMM);
        k_prep_base<<<cdiv_(n, 256), 256, 0, stream>>>(x, bb, n);
        int Kp = packw(stem_base, 32, 3 * 49);
        conv(bb, ws + oA, 3, 224, 224, 32, 112, 112, 7, 2, 3, 3 * 49, Kp, 0);
        k_prep_poly<<<cdiv_(n, 256), 256, 0, stream>>>(x, pb, ws + oMM, 4, 3, 224 * 224);
        Kp = packw(stem_poly, 32, 12 * 49);
        conv(pb, ws + oA, 12, 224, 224, 32, 112, 112, 7, 2, 3, 12 * 49, Kp, 1);
        k_instnorm<<<4 * 32, 256, 0, stream>>>(ws + oA, 112 * 112);
        int np = 4 * 32 * 56 * 56;
        k_maxpool<<<cdiv_(np, 256), 256, 0, stream>>>(ws + oA, ws + oB, 4 * 32, 112, 112, 56, 56);
    }

    float* hcur = ws + oB;
    float* tbuf = ws + oA;

    const int cins[4]    = {32, 16, 32, 64};
    const int couts[4]   = {16, 32, 64, 128};
    const int strides[4] = {1, 2, 2, 2};
    const int Hs[4]      = {56, 56, 28, 14};
    const int pidx[4]    = {3, 10, 17, 24};

    for (int l = 0; l < 4; ++l) {
        const int Cin = cins[l], Cout = couts[l], S = strides[l];
        const int H = Hs[l], W = H;
        const int OH = (H - 1) / S + 1, OW = OH;
        const float* moe_base = (const float*)d_in[pidx[l] + 0];
        const float* moe_poly = (const float*)d_in[pidx[l] + 1];
        const float* w_gate   = (const float*)d_in[pidx[l] + 2];
        const float* c2b      = (const float*)d_in[pidx[l] + 3];
        const float* c2p      = (const float*)d_in[pidx[l] + 4];
        const float* dsb      = (const float*)d_in[pidx[l] + 5];
        const float* dsp      = (const float*)d_in[pidx[l] + 6];
        const int nin = 4 * Cin * H * W;
        const int ysz = 4 * Cout * OH * OW;

        // gating
        k_chan_mean<<<4 * Cin, 256, 0, stream>>>(hcur, ws + oGI, H * W);
        k_gate<<<1, 32, 0, stream>>>(ws + oGI, w_gate, Cin, ws + oGT, loss);
        k_expert_minmax<<<8, 256, 0, stream>>>(hcur, ws + oGT, 4, Cin * H * W, ws + oMM + 2);

        k_zero<<<cdiv_(ysz, 256), 256, 0, stream>>>(ws + oY, ysz);
        // SiLU(x) in bf16 is expert-independent: compute once per layer
        k_prep_base<<<cdiv_(nin, 256), 256, 0, stream>>>(hcur, bb, nin);

        // dense dispatch: all 8 experts on full batch, gated combine
        for (int e = 0; e < 8; ++e) {
            const float* bw = moe_base + (size_t)e * Cout * Cin * 9;
            const float* pw = moe_poly + (size_t)e * Cout * Cin * 4 * 9;
            int Kp = packw(bw, Cout, Cin * 9);
            conv(bb, tbuf, Cin, H, W, Cout, OH, OW, 3, S, 1, Cin * 9, Kp, 0);
            k_prep_poly<<<cdiv_(nin, 256), 256, 0, stream>>>(hcur, pb, ws + oMM + 2 + 2 * e,
                                                             4, Cin, H * W);
            Kp = packw(pw, Cout, Cin * 4 * 9);
            conv(pb, tbuf, Cin * 4, H, W, Cout, OH, OW, 3, S, 1, Cin * 4 * 9, Kp, 1);
            k_instnorm<<<4 * Cout, 256, 0, stream>>>(tbuf, OH * OW);
            k_combine<<<cdiv_(ysz, 256), 256, 0, stream>>>(ws + oY, tbuf, ws + oGT, e,
                                                           Cout * OH * OW, ysz);
        }

        // conv2: KALN 1x1 s1 on MoE output
        k_global_minmax<<<1, 1024, 0, stream>>>(ws + oY, ysz, ws + oMM);
        k_prep_base<<<cdiv_(ysz, 256), 256, 0, stream>>>(ws + oY, bb, ysz);
        int Kp = packw(c2b, Cout, Cout);
        conv(bb, ws + oC2, Cout, OH, OW, Cout, OH, OW, 1, 1, 0, Cout, Kp, 0);
        k_prep_poly<<<cdiv_(ysz, 256), 256, 0, stream>>>(ws + oY, pb, ws + oMM, 4, Cout, OH * OW);
        Kp = packw(c2p, Cout, Cout * 4);
        conv(pb, ws + oC2, Cout * 4, OH, OW, Cout, OH, OW, 1, 1, 0, Cout * 4, Kp, 1);
        k_instnorm<<<4 * Cout, 256, 0, stream>>>(ws + oC2, OH * OW);

        // downsample: KALN 1x1 stride S on block input
        k_global_minmax<<<1, 1024, 0, stream>>>(hcur, nin, ws + oMM);
        k_prep_base<<<cdiv_(nin, 256), 256, 0, stream>>>(hcur, bb, nin);
        Kp = packw(dsb, Cout, Cin);
        conv(bb, ws + oDS, Cin, H, W, Cout, OH, OW, 1, S, 0, Cin, Kp, 0);
        k_prep_poly<<<cdiv_(nin, 256), 256, 0, stream>>>(hcur, pb, ws + oMM, 4, Cin, H * W);
        Kp = packw(dsp, Cout, Cin * 4);
        conv(pb, ws + oDS, Cin * 4, H, W, Cout, OH, OW, 1, S, 0, Cin * 4, Kp, 1);
        k_instnorm<<<4 * Cout, 256, 0, stream>>>(ws + oDS, OH * OW);

        // residual add -> tbuf, swap buffers
        k_add<<<cdiv_(ysz, 256), 256, 0, stream>>>(ws + oC2, ws + oDS, tbuf, ysz);
        float* tmp = hcur; hcur = tbuf; tbuf = tmp;
    }

    // head: global average pool + FC
    k_chan_mean<<<4 * 128, 256, 0, stream>>>(hcur, ws + oFT, 7 * 7);
    k_fc<<<cdiv_(4000, 256), 256, 0, stream>>>(ws + oFT, fc_w, fc_b, (float*)d_out);
}
